// CrossAttention_34832184771360
// MI455X (gfx1250) — matrix operand order repaired and checked by reading, not compiled, not run on hardware
//
#include <hip/hip_runtime.h>
#include <hip/hip_bf16.h>
#include <cstdint>

// Problem constants (from reference)
#define H_    12
#define Bsz   8
#define Nseq  4096
#define Cdim  768
#define Dh    64
#define NB    512
#define Mrows (Bsz*Nseq)   // 32768

typedef __attribute__((ext_vector_type(16))) __bf16 v16bf;
typedef __attribute__((ext_vector_type(8)))  float  v8f;
typedef unsigned short u16;

union Frag  { v16bf v; u16 u[16]; uint4 q[2]; };
union CFrag { v8f v; float f[8]; };

__device__ __forceinline__ u16 f2bf(float f) {
    unsigned int u = __float_as_uint(f);
    u += 0x7FFFu + ((u >> 16) & 1u);          // round-to-nearest-even
    return (u16)(u >> 16);
}
__device__ __forceinline__ float bf2f(u16 x) {
    return __uint_as_float(((unsigned int)x) << 16);
}
__device__ __forceinline__ v8f wmma_bf16(v16bf a, v16bf b, v8f c) {
    // 8 args: (neg_a, A, neg_b, B, c_mod, C, reuse_a, reuse_b)
    return __builtin_amdgcn_wmma_f32_16x16x32_bf16(false, a, false, b, (short)0, c, false, false);
}
__device__ __forceinline__ v8f vzero8() {
    v8f z = {0.f,0.f,0.f,0.f,0.f,0.f,0.f,0.f};
    return z;
}

// ---------------------------------------------------------------------------
// Prep 0: x -> bf16 (one streaming pass; removes per-head re-read of fp32 x
// and strips cvt VALU from the GEMM hot loop)
// ---------------------------------------------------------------------------
__global__ void prep_xbf(const float* __restrict__ x, u16* __restrict__ xb) {
    int i = blockIdx.x*256 + threadIdx.x;          // quad index
    float4 f = ((const float4*)x)[i];
    uint2 o;
    o.x = (unsigned)f2bf(f.x) | ((unsigned)f2bf(f.y) << 16);
    o.y = (unsigned)f2bf(f.z) | ((unsigned)f2bf(f.w) << 16);
    ((uint2*)xb)[i] = o;
}

// ---------------------------------------------------------------------------
// Prep 1: bf16 transposes of Wq, Wlin (B operand wants N-major, K-contiguous)
// ---------------------------------------------------------------------------
__global__ void prep_wt(const float* __restrict__ Wq, const float* __restrict__ Wlin,
                        u16* __restrict__ WqT, u16* __restrict__ WlinT) {
    int i = blockIdx.x * 256 + threadIdx.x;         // [0, 2*768*768)
    const float* src = Wq; u16* dst = WqT;
    if (i >= Cdim*Cdim) { i -= Cdim*Cdim; src = Wlin; dst = WlinT; }
    int n = i / Cdim, k = i - n*Cdim;
    dst[n*Cdim + k] = f2bf(src[k*Cdim + n]);
}

// ---------------------------------------------------------------------------
// Prep 2: kv = bank @ Wkv ; k raw fp32 (H,Nb,d), v transposed bf16 (H,d,Nb)
// ---------------------------------------------------------------------------
__global__ void prep_kv(const float* __restrict__ bank, const float* __restrict__ Wkv,
                        float* __restrict__ kraw, u16* __restrict__ vT) {
    int i = blockIdx.x*256 + threadIdx.x;           // [0, 512*1536)
    int m = i / 1536; int j = i - m*1536;
    float s = 0.f;
    #pragma unroll
    for (int f = 0; f < 16; ++f) s += bank[m*16 + f] * Wkv[f*1536 + j];
    int which = j / 768;
    int jj = j - which*768;
    int h = jj >> 6, dd = jj & 63;
    if (which == 0) kraw[((h<<9) + m)*Dh + dd] = s;       // k: (H, Nb, d)
    else            vT[((h<<6) + dd)*NB + m]   = f2bf(s); // vT: (H, d, Nb)
}

// ---------------------------------------------------------------------------
// Prep 3: L2-normalize k rows -> bf16
// ---------------------------------------------------------------------------
__global__ void prep_knorm(const float* __restrict__ kraw, u16* __restrict__ kbf) {
    int r = blockIdx.x*256 + threadIdx.x;           // [0, 12*512)
    if (r >= H_*NB) return;
    const float* src = kraw + (size_t)r*Dh;
    float s = 0.f;
    #pragma unroll 8
    for (int d = 0; d < Dh; ++d) { float v = src[d]; s += v*v; }
    float inv = rsqrtf(fmaxf(s, 1e-24f));
    #pragma unroll 8
    for (int d = 0; d < Dh; ++d) kbf[(size_t)r*Dh + d] = f2bf(src[d]*inv);
}

// ---------------------------------------------------------------------------
// GEMM1: q = l2norm_perhead(xb @ Wq), bf16 in/out, software-pipelined k-loop.
// block = 128 (4 waves); wave -> 16 rows x one head's 64 cols (4 n-tiles)
// ---------------------------------------------------------------------------
__global__ void __launch_bounds__(128) gemm_q(const u16* __restrict__ xb,
                                              const u16* __restrict__ WqT,
                                              u16* __restrict__ qout) {
    __shared__ u16 qst[4][16*Dh];                   // per-wave coalescing tile
    const int wave = threadIdx.x >> 5;
    const int lane = threadIdx.x & 31;
    const int h    = blockIdx.y;
    const int rowbase = (blockIdx.x*4 + wave) * 16;
    const int mlo   = lane & 15;
    const int khalf = (lane >> 4) << 3;             // A-layout K split
    const int kb16  = (lane >> 4) << 3;             // B-layout K base
    CFrag acc[4];
    #pragma unroll
    for (int t = 0; t < 4; ++t) acc[t].v = vzero8();

    const u16* arow = xb  + (size_t)(rowbase + mlo)*Cdim;
    const u16* brow = WqT + (size_t)(h*Dh + mlo)*Cdim;
    Frag a[2]; Frag b[2][4];
    auto LD = [&](int buf, int k0) {
        const u16* xr = arow + k0 + khalf;
        a[buf].q[0] = *(const uint4*)(xr);
        a[buf].q[1] = *(const uint4*)(xr + 16);
        #pragma unroll
        for (int t = 0; t < 4; ++t) {
            const u16* wr = brow + t*16*Cdim + k0 + kb16;
            b[buf][t].q[0] = *(const uint4*)(wr);
            b[buf][t].q[1] = *(const uint4*)(wr + 16);
        }
    };
    LD(0, 0);
    #pragma unroll
    for (int it = 0; it < 24; ++it) {               // double-buffered pipeline
        const int cb = it & 1;
        if (it + 1 < 24) LD(cb ^ 1, (it + 1) * 32);
        #pragma unroll
        for (int t = 0; t < 4; ++t)
            acc[t].v = wmma_bf16(a[cb].v, b[cb][t].v, acc[t].v);
    }

    // fused per-head L2 normalize: reduce sum of squares across 16 lanes / row
    #pragma unroll
    for (int r = 0; r < 8; ++r) {
        float v0 = acc[0].f[r], v1 = acc[1].f[r], v2 = acc[2].f[r], v3 = acc[3].f[r];
        float s = v0*v0 + v1*v1 + v2*v2 + v3*v3;
        s += __shfl_xor(s, 1, 32); s += __shfl_xor(s, 2, 32);
        s += __shfl_xor(s, 4, 32); s += __shfl_xor(s, 8, 32);
        float inv = rsqrtf(fmaxf(s, 1e-24f));
        int m = r + ((lane >> 4) << 3);
        u16* row = &qst[wave][m*Dh + mlo];
        row[0]  = f2bf(v0*inv);
        row[16] = f2bf(v1*inv);
        row[32] = f2bf(v2*inv);
        row[48] = f2bf(v3*inv);
    }
    __syncthreads();
    // coalesced bf16 writeout: q layout (B,H,N,d)
    int lrow = lane >> 1, half = lane & 1;
    int rowg = rowbase + lrow;
    int bidx = rowg >> 12, n = rowg & 4095;
    const uint4* srcq = (const uint4*)&qst[wave][lrow*Dh + half*32];
    uint4* dstq = (uint4*)(qout + (((size_t)bidx*H_ + h)*Nseq + n)*Dh + half*32);
    dstq[0] = srcq[0]; dstq[1] = srcq[1]; dstq[2] = srcq[2]; dstq[3] = srcq[3];
}

// ---------------------------------------------------------------------------
// Fused attention: per block = one (b,h) and 64 query rows (4 waves x 16).
// k (Nbxd) and vT (dxNb) staged in LDS with async DMA (ASYNCcnt path);
// scores + softmax + PV all in LDS.
// ---------------------------------------------------------------------------
__global__ void __launch_bounds__(128) attn_kernel(const u16* __restrict__ qin,
                                                   const u16* __restrict__ kin,
                                                   const u16* __restrict__ vTin,
                                                   const float* __restrict__ temperature,
                                                   u16* __restrict__ ctx) {
    __shared__ u16 kS[NB*Dh];        // 64 KB
    __shared__ u16 vS[Dh*NB];        // 64 KB (v transposed)
    __shared__ u16 pS[4][16*NB];     // 64 KB scores->probs, per wave
    __shared__ float invS[4][16];

    const int tid  = threadIdx.x;
    const int wave = tid >> 5, lane = tid & 31;
    const int bh = blockIdx.y;
    const int b = bh / H_, h = bh - b*H_;

    {   // async DMA stage of k and vT into LDS (GLOBAL_LOAD_ASYNC_TO_LDS_B128)
        unsigned ldsK = (unsigned)(uintptr_t)(void*)&kS[0];
        unsigned ldsV = (unsigned)(uintptr_t)(void*)&vS[0];
        unsigned long long kg = (unsigned long long)(uintptr_t)(kin  + (size_t)h*NB*Dh);
        unsigned long long vg = (unsigned long long)(uintptr_t)(vTin + (size_t)h*Dh*NB);
        #pragma unroll 1
        for (int i = tid*16; i < NB*Dh*2; i += 128*16) {   // 64 KB each, 16B chunks
            asm volatile("global_load_async_to_lds_b128 %0, %1, %2"
                         :: "v"(ldsK + i), "v"(i), "s"(kg) : "memory");
            asm volatile("global_load_async_to_lds_b128 %0, %1, %2"
                         :: "v"(ldsV + i), "v"(i), "s"(vg) : "memory");
        }
        asm volatile("s_wait_asynccnt 0x0" ::: "memory");
    }
    __syncthreads();

    const int mlo   = lane & 15;
    const int khalf = (lane >> 4) << 3;   // A layout
    const int kb16  = (lane >> 4) << 3;   // B layout
    const int row0  = blockIdx.x*64 + wave*16;

    // q A-fragments for this wave's 16 rows (K = d = 64 -> two 16x32 tiles)
    Frag a0, a1;
    {
        const u16* qr = qin + (((size_t)b*H_ + h)*Nseq + row0 + mlo)*Dh;
        a0.q[0] = *(const uint4*)(qr + khalf);
        a0.q[1] = *(const uint4*)(qr + 16 + khalf);
        a1.q[0] = *(const uint4*)(qr + 32 + khalf);
        a1.q[1] = *(const uint4*)(qr + 48 + khalf);
    }
    const float temp = temperature[h];
    u16* P = pS[wave];

    // ---- scores: S = (q . k^T) * temp, stored bf16 in LDS (16 x 512) ----
    for (int nt = 0; nt < 32; ++nt) {
        Frag b0, b1;
        const u16* kr = kS + (nt*16 + mlo)*Dh;
        b0.q[0] = *(const uint4*)(kr + kb16);
        b0.q[1] = *(const uint4*)(kr + kb16 + 16);
        b1.q[0] = *(const uint4*)(kr + 32 + kb16);
        b1.q[1] = *(const uint4*)(kr + 32 + kb16 + 16);
        CFrag s; s.v = vzero8();
        s.v = wmma_bf16(a0.v, b0.v, s.v);
        s.v = wmma_bf16(a1.v, b1.v, s.v);
        #pragma unroll
        for (int r = 0; r < 8; ++r) {
            int m = r + ((lane >> 4) << 3);
            P[m*NB + nt*16 + mlo] = f2bf(s.f[r] * temp);
        }
    }

    // ---- in-place softmax (wave-local): 2 lanes per row, 256 cols each ----
    {
        int srow = lane >> 1, half = lane & 1;
        uint4* pr = (uint4*)(P + srow*NB + half*256);
        float mx = -1e30f;
        #pragma unroll 4
        for (int i = 0; i < 32; ++i) {
            uint4 u = pr[i];
            const u16* e = (const u16*)&u;
            #pragma unroll
            for (int j = 0; j < 8; ++j) mx = fmaxf(mx, bf2f(e[j]));
        }
        mx = fmaxf(mx, __shfl_xor(mx, 1, 32));
        float sum = 0.f;
        #pragma unroll 4
        for (int i = 0; i < 32; ++i) {
            uint4 u = pr[i];
            u16* e = (u16*)&u;
            #pragma unroll
            for (int j = 0; j < 8; ++j) {
                float p = __expf(bf2f(e[j]) - mx);
                sum += p;
                e[j] = f2bf(p);
            }
            pr[i] = u;
        }
        sum += __shfl_xor(sum, 1, 32);
        if (half == 0) invS[wave][srow] = 1.0f / sum;  // row scale folded into PV
    }

    // ---- out = P @ v : K = 512 (16 k-tiles), 4 d-tiles of 16 cols ----
    #pragma unroll
    for (int dt = 0; dt < 4; ++dt) {
        CFrag o; o.v = vzero8();
        for (int kt = 0; kt < 16; ++kt) {
            Frag a, bb;
            const u16* prw = P + mlo*NB + kt*32 + khalf;
            a.q[0] = *(const uint4*)(prw);
            a.q[1] = *(const uint4*)(prw + 16);
            const u16* vr = vS + (dt*16 + mlo)*NB + kt*32 + kb16;
            bb.q[0] = *(const uint4*)(vr);
            bb.q[1] = *(const uint4*)(vr + 16);
            o.v = wmma_bf16(a.v, bb.v, o.v);
        }
        #pragma unroll
        for (int r = 0; r < 8; ++r) {
            int m = r + ((lane >> 4) << 3);
            float val = o.f[r] * invS[wave][m];
            ctx[((size_t)b*Nseq + row0 + m)*Cdim + h*Dh + dt*16 + mlo] = f2bf(val);
        }
    }
}

// ---------------------------------------------------------------------------
// GEMM2: out = PReLU(ctx @ Wlin + b_lin), fp32 output, software-pipelined
// ---------------------------------------------------------------------------
__global__ void __launch_bounds__(128) gemm_out(const u16* __restrict__ ctx,
                                                const u16* __restrict__ WlinT,
                                                const float* __restrict__ b_lin,
                                                const float* __restrict__ prelu_a,
                                                float* __restrict__ out) {
    const int wave = threadIdx.x >> 5, lane = threadIdx.x & 31;
    const int ng = blockIdx.y;
    const int rowbase = (blockIdx.x*4 + wave) * 16;
    const int mlo   = lane & 15;
    const int khalf = (lane >> 4) << 3;
    const int kb16  = (lane >> 4) << 3;
    CFrag acc[4];
    #pragma unroll
    for (int t = 0; t < 4; ++t) acc[t].v = vzero8();

    const u16* arow = ctx   + (size_t)(rowbase + mlo)*Cdim;
    const u16* brow = WlinT + (size_t)(ng*64 + mlo)*Cdim;
    Frag a[2]; Frag b[2][4];
    auto LD = [&](int buf, int k0) {
        const u16* xr = arow + k0 + khalf;
        a[buf].q[0] = *(const uint4*)(xr);
        a[buf].q[1] = *(const uint4*)(xr + 16);
        #pragma unroll
        for (int t = 0; t < 4; ++t) {
            const u16* wr = brow + t*16*Cdim + k0 + kb16;
            b[buf][t].q[0] = *(const uint4*)(wr);
            b[buf][t].q[1] = *(const uint4*)(wr + 16);
        }
    };
    LD(0, 0);
    #pragma unroll
    for (int it = 0; it < 24; ++it) {               // double-buffered pipeline
        const int cb = it & 1;
        if (it + 1 < 24) LD(cb ^ 1, (it + 1) * 32);
        #pragma unroll
        for (int t = 0; t < 4; ++t)
            acc[t].v = wmma_bf16(a[cb].v, b[cb][t].v, acc[t].v);
    }

    float pa = prelu_a[0];
    #pragma unroll
    for (int t = 0; t < 4; ++t) {
        int n = ng*64 + t*16 + mlo;
        float bias = b_lin[n];
        #pragma unroll
        for (int r = 0; r < 8; ++r) {
            int m = r + ((lane >> 4) << 3);
            float v = acc[t].f[r] + bias;
            v = (v >= 0.f) ? v : pa*v;
            out[(size_t)(rowbase + m)*Cdim + n] = v;
        }
    }
}

// ---------------------------------------------------------------------------
extern "C" void kernel_launch(void* const* d_in, const int* in_sizes, int n_in,
                              void* d_out, int out_size, void* d_ws, size_t ws_size,
                              hipStream_t stream) {
    const float* x     = (const float*)d_in[0];
    const float* bank  = (const float*)d_in[1];
    const float* Wq    = (const float*)d_in[2];
    const float* Wkv   = (const float*)d_in[3];
    const float* Wlin  = (const float*)d_in[4];
    const float* b_lin = (const float*)d_in[5];
    const float* pa    = (const float*)d_in[6];
    const float* temp  = (const float*)d_in[7];
    float* out = (float*)d_out;

    char* ws = (char*)d_ws;
    u16*   WqT   = (u16*)  (ws + 0);          //  768*768*2  = 1,179,648
    u16*   WlinT = (u16*)  (ws + 1179648);    //  1,179,648
    float* kraw  = (float*)(ws + 2359296);    //  12*512*64*4 = 1,572,864
    u16*   kbf   = (u16*)  (ws + 3932160);    //  786,432
    u16*   vT    = (u16*)  (ws + 4718592);    //  786,432
    u16*   q     = (u16*)  (ws + 5505024);    //  8*12*4096*64*2 = 50,331,648
    u16*   ctx   = (u16*)  (ws + 55836672);   //  8*4096*768*2   = 50,331,648
    u16*   xbf   = ctx;   // alias: xbf live only before attn writes ctx
    // total workspace: ~106.2 MB

    prep_xbf  <<<24576, 256, 0, stream>>>(x, xbf);
    prep_wt   <<<4608, 256, 0, stream>>>(Wq, Wlin, WqT, WlinT);
    prep_kv   <<<3072, 256, 0, stream>>>(bank, Wkv, kraw, vT);
    prep_knorm<<<24,   256, 0, stream>>>(kraw, kbf);
    gemm_q    <<<dim3(512, 12), 128, 0, stream>>>(xbf, WqT, q);
    attn_kernel<<<dim3(Nseq/64, Bsz*H_), 128, 0, stream>>>(q, kbf, vT, temp, ctx);
    gemm_out  <<<dim3(512, 12), 128, 0, stream>>>(ctx, WlinT, b_lin, pa, out);
}
